// TFEfficientFormerSelfAttention_47253230191065
// MI455X (gfx1250) — matrix operand order repaired and checked by reading, not compiled, not run on hardware
//
#include <hip/hip_runtime.h>
#include <hip/hip_bf16.h>

typedef unsigned short u16;
typedef unsigned int   u32;
typedef __attribute__((ext_vector_type(16))) __bf16 v16bf;
typedef __attribute__((ext_vector_type(8)))  float   v8f;
typedef __attribute__((ext_vector_type(16))) u16     v16u;
typedef __attribute__((ext_vector_type(8)))  u16     v8u;
typedef __attribute__((ext_vector_type(4)))  int     v4i;

#define NUM_HEADS_C   8
#define SEQ_C         256
#define QKV_DIM_C     1536
#define HEAD_STRIDE_C 192      // 32 q + 32 k + 128 v
#define TOTAL_EXP_C   1024
#define SCALE_C       0.17677669529663687f   // 32^-0.5

#define SST 260   // fp32 stride for softmax rows (bank-conflict pad)
#define VST 264   // u16 stride for transposed V rows

// ---- CDNA5 async global->LDS path (guarded) ------------------------------
#if defined(__has_builtin)
# if __has_builtin(__builtin_amdgcn_global_load_async_to_lds_b128)
#  define HAVE_ASYNC_LDS 1
# endif
#endif
#ifndef HAVE_ASYNC_LDS
# define HAVE_ASYNC_LDS 0
#endif

__device__ __forceinline__ void async_cp16(const void* g, void* l) {
#if HAVE_ASYNC_LDS
    __builtin_amdgcn_global_load_async_to_lds_b128(
        (v4i*)(uintptr_t)g, (v4i*)(uintptr_t)l, 0, 0);
#else
    *(v8u*)l = *(const v8u*)g;
#endif
}

__device__ __forceinline__ void wait_async0() {
#if HAVE_ASYNC_LDS
# if defined(__has_builtin) && __has_builtin(__builtin_amdgcn_s_wait_asynccnt)
    __builtin_amdgcn_s_wait_asynccnt(0);
# else
    asm volatile("s_wait_asynccnt 0" ::: "memory");
# endif
#endif
}

// ---- helpers -------------------------------------------------------------

__device__ __forceinline__ u16 f2bf(float x) {
    u32 u = __builtin_bit_cast(u32, x);
    u32 r = u + 0x7FFFu + ((u >> 16) & 1u);     // RNE
    return (u16)(r >> 16);
}
__device__ __forceinline__ u32 pack2bf(float a, float b) {
    return (u32)f2bf(a) | ((u32)f2bf(b) << 16);
}

// A-operand fragment: 16x32 bf16, rows striped over lane%16,
// lanes 0-15: K 0..7 then 16..23 ; lanes 16-31: K 8..15 then 24..31.
__device__ __forceinline__ v16bf load_fragA_k32(const u16* base, int stride) {
    int lane = threadIdx.x & 31;
    const u16* p = base + (lane & 15) * stride + ((lane >> 4) << 3);
    v8u lo = *(const v8u*)(p);
    v8u hi = *(const v8u*)(p + 16);
    v16u t;
    #pragma unroll
    for (int i = 0; i < 8; ++i) { t[i] = lo[i]; t[i + 8] = hi[i]; }
    return __builtin_bit_cast(v16bf, t);
}

// B-operand fragment: 32x16 bf16 stored as [N][K] in LDS.
// lanes 0-15: K 0..7 then 16..23 ; lanes 16-31: K 8..15 then 24..31.
__device__ __forceinline__ v16bf load_fragB_k32(const u16* base, int stride) {
    int lane = threadIdx.x & 31;
    const u16* p = base + (lane & 15) * stride + ((lane >> 4) << 3);
    v8u lo = *(const v8u*)(p);
    v8u hi = *(const v8u*)(p + 16);
    v16u t;
    #pragma unroll
    for (int i = 0; i < 8; ++i) { t[i] = lo[i]; t[i + 8] = hi[i]; }
    return __builtin_bit_cast(v16bf, t);
}

__device__ __forceinline__ v8f wmma_bf16(v16bf a, v16bf b, v8f c) {
    return __builtin_amdgcn_wmma_f32_16x16x32_bf16(
        false, a, false, b, (short)0, c, false, false);
}

// ---- kernel 1: positional-bias gather ------------------------------------

__global__ __launch_bounds__(256) void bias_gather_kernel(
    const float* __restrict__ biases,   // [8][256]
    const int*   __restrict__ idxs,     // [256*256]
    float*       __restrict__ out)      // [8][256*256]
{
    int o  = blockIdx.x * 256 + threadIdx.x;
    int h  = o >> 16;
    int qk = o & 65535;
    out[o] = biases[h * 256 + idxs[qk]];
}

// ---- kernels 2 & 4: bf16 WMMA GEMM  C = A*W + bias -----------------------
// A: [M][K] (f32 or bf16), W: [K][N] f32, C: [M][N] (f32 or bf16)
// 128x128 macro-tile, BK=64, 256 threads = 8 waves (4M x 2N),
// each wave owns 32x64 => 16 WMMAs per barrier pair.

template <bool A_BF16, bool OUT_BF16>
__global__ __launch_bounds__(256) void gemm_bf16_kernel(
    const void*  __restrict__ Ain,
    const float* __restrict__ W,
    const float* __restrict__ bias,
    void*        __restrict__ Cout,
    int M, int N, int K)
{
    __shared__ u16 la[128 * 64];   // [M][K] bf16
    __shared__ u16 lb[128 * 64];   // [N][K] bf16 (transposed W tile)

    const int tid  = threadIdx.x;
    const int wave = tid >> 5;
    const int m0   = blockIdx.x * 128;
    const int n0   = blockIdx.y * 128;
    const int wm   = (wave >> 1) * 32;
    const int wn   = (wave & 1) * 64;

    v8f acc[2][4] = {};

    const int ksteps = K >> 6;
    for (int kt = 0; kt < ksteps; ++kt) {
        const int k0 = kt << 6;

        // ---- A tile 128x64 -> la ----
        if (A_BF16) {
            // 1024 chunks of 8 bf16, 4 per thread; async direct to LDS
            #pragma unroll
            for (int i = 0; i < 4; ++i) {
                int chunk = i * 256 + tid;
                int kk8 = chunk & 7, mm = chunk >> 3;
                const u16* gp = (const u16*)Ain + (size_t)(m0 + mm) * K + k0 + kk8 * 8;
                async_cp16(gp, &la[mm * 64 + kk8 * 8]);
            }
        } else {
            // 2048 float4 chunks, 8 per thread; convert + packed stores
            #pragma unroll
            for (int i = 0; i < 8; ++i) {
                int chunk = i * 256 + tid;
                int kk4 = chunk & 15, mm = chunk >> 4;
                float4 v = *(const float4*)((const float*)Ain +
                              (size_t)(m0 + mm) * K + k0 + kk4 * 4);
                u32* lp = (u32*)&la[mm * 64 + kk4 * 4];
                lp[0] = pack2bf(v.x, v.y);
                lp[1] = pack2bf(v.z, v.w);
            }
        }

        // ---- W tile 64x128 -> lb[n][k] (b128 loads, transposed stores) ----
        #pragma unroll
        for (int i = 0; i < 8; ++i) {
            int chunk = i * 256 + tid;
            int n4 = chunk & 31, kk = chunk >> 5;
            int gn = n0 + n4 * 4;
            float x0 = 0.f, x1 = 0.f, x2 = 0.f, x3 = 0.f;
            if (gn < N) {   // N is a multiple of 4: whole-float4 guard
                float4 v = *(const float4*)&W[(size_t)(k0 + kk) * N + gn];
                x0 = v.x; x1 = v.y; x2 = v.z; x3 = v.w;
            }
            lb[(n4 * 4 + 0) * 64 + kk] = f2bf(x0);
            lb[(n4 * 4 + 1) * 64 + kk] = f2bf(x1);
            lb[(n4 * 4 + 2) * 64 + kk] = f2bf(x2);
            lb[(n4 * 4 + 3) * 64 + kk] = f2bf(x3);
        }

        if (A_BF16) wait_async0();
        __syncthreads();

        #pragma unroll
        for (int ks = 0; ks < 2; ++ks) {
            v16bf af[2], bfr[4];
            #pragma unroll
            for (int i = 0; i < 2; ++i)
                af[i] = load_fragA_k32(&la[(wm + i * 16) * 64 + ks * 32], 64);
            #pragma unroll
            for (int j = 0; j < 4; ++j)
                bfr[j] = load_fragB_k32(&lb[(wn + j * 16) * 64 + ks * 32], 64);
            #pragma unroll
            for (int i = 0; i < 2; ++i)
                #pragma unroll
                for (int j = 0; j < 4; ++j)
                    acc[i][j] = wmma_bf16(af[i], bfr[j], acc[i][j]);
        }
        __syncthreads();
    }

    // epilogue: C layout N=lane%16, M=vgpr + 8*(lane>=16)
    const int lane = tid & 31;
    const int nl = lane & 15;
    const int mh = (lane >> 4) << 3;
    #pragma unroll
    for (int i = 0; i < 2; ++i) {
        #pragma unroll
        for (int j = 0; j < 4; ++j) {
            int n = n0 + wn + j * 16 + nl;
            if (n >= N) continue;
            float bv = bias[n];
            #pragma unroll
            for (int r = 0; r < 8; ++r) {
                int m = m0 + wm + i * 16 + r + mh;
                float val = acc[i][j][r] + bv;
                if (OUT_BF16) ((u16*)Cout)[(size_t)m * N + n] = f2bf(val);
                else          ((float*)Cout)[(size_t)m * N + n] = val;
            }
        }
    }
}

// ---- kernel 3: fused attention (per batch, head, 128-query block) --------

__global__ __launch_bounds__(256) void attn_kernel(
    const u16*   __restrict__ qkv,      // [B*SEQ][1536] bf16
    const float* __restrict__ biasmat,  // [8][256][256] f32
    u16*         __restrict__ ctx)      // [B*SEQ][1024] bf16
{
    extern __shared__ char smem[];
    u16*   lq   = (u16*)smem;                 // [128][32]
    u16*   lk   = lq + 128 * 32;              // [256][32]
    u16*   lv   = lk + 256 * 32;              // [128][VST]  ([d][s])
    float* ls   = (float*)(lv + 128 * VST);   // [8 waves][16][SST]
    float* linv = ls + 8 * 16 * SST;          // [8][16]

    const int tid  = threadIdx.x;
    const int lane = tid & 31;
    const int wave = tid >> 5;
    const int bid  = blockIdx.x;
    const int mb   = bid & 1;
    const int h    = (bid >> 1) & 7;
    const int b    = bid >> 4;
    const size_t row0 = (size_t)b * SEQ_C * QKV_DIM_C;
    const int hq = h * HEAD_STRIDE_C;

    // q: 128x32 bf16 = 512 16B chunks (async -> LDS)
    #pragma unroll
    for (int i = 0; i < 2; ++i) {
        int chunk = i * 256 + tid;
        int m = chunk >> 2, kk8 = chunk & 3;
        const u16* gp = qkv + row0 + (size_t)(mb * 128 + m) * QKV_DIM_C + hq + kk8 * 8;
        async_cp16(gp, &lq[m * 32 + kk8 * 8]);
    }
    // k: 256x32 bf16 = 1024 16B chunks (async -> LDS)
    #pragma unroll
    for (int i = 0; i < 4; ++i) {
        int chunk = i * 256 + tid;
        int s = chunk >> 2, kk8 = chunk & 3;
        const u16* gp = qkv + row0 + (size_t)s * QKV_DIM_C + hq + 32 + kk8 * 8;
        async_cp16(gp, &lk[s * 32 + kk8 * 8]);
    }
    // v: 256x128 bf16, b128 loads + transposed stores -> lv[d][s]
    #pragma unroll
    for (int i = 0; i < 16; ++i) {
        int chunk = i * 256 + tid;          // 4096 chunks of 8
        int d8 = chunk & 15, s = chunk >> 4;
        v8u v = *(const v8u*)(qkv + row0 + (size_t)s * QKV_DIM_C + hq + 64 + d8 * 8);
        #pragma unroll
        for (int e = 0; e < 8; ++e)
            lv[(d8 * 8 + e) * VST + s] = v[e];
    }
    wait_async0();
    __syncthreads();

    // S = q k^T * scale + bias ; wave owns 16 query rows
    const int mw = wave * 16;
    float* Sw = ls + wave * 16 * SST;
    const int nl = lane & 15;
    const int mh = (lane >> 4) << 3;
    const float* biash = biasmat + (size_t)h * 65536;

    v16bf aq = load_fragA_k32(&lq[mw * 32], 32);
    #pragma unroll
    for (int j = 0; j < 16; ++j) {
        v16bf bk = load_fragB_k32(&lk[j * 16 * 32], 32);
        v8f c = {};
        c = wmma_bf16(aq, bk, c);
        int kcol = j * 16 + nl;
        #pragma unroll
        for (int r = 0; r < 8; ++r) {
            int m = r + mh;
            float sv = c[r] * SCALE_C
                     + biash[(size_t)(mb * 128 + mw + m) * 256 + kcol];
            Sw[m * SST + kcol] = sv;
        }
    }
    asm volatile("s_wait_dscnt 0" ::: "memory");  // wave-local LDS RAW

    // row softmax: lane -> row lane&15, column half lane>>4
    {
        int r = lane & 15, half = lane >> 4;
        float* row = Sw + r * SST + half * 128;
        float mx = -1e30f;
        for (int c = 0; c < 128; ++c) mx = fmaxf(mx, row[c]);
        mx = fmaxf(mx, __shfl_xor(mx, 16, 32));
        float sum = 0.0f;
        for (int c = 0; c < 128; ++c) {
            float p = __expf(row[c] - mx);
            row[c] = p;
            sum += p;
        }
        sum += __shfl_xor(sum, 16, 32);
        if (half == 0) linv[wave * 16 + r] = 1.0f / sum;
    }
    asm volatile("s_wait_dscnt 0" ::: "memory");

    // O = P V : wave computes 16 x 128 ; P rebuilt as bf16 A-fragments
    v8f o[8] = {};
    for (int kb = 0; kb < 8; ++kb) {
        int m = lane & 15;
        int kbase = kb * 32 + ((lane >> 4) << 3);
        const float* srow = Sw + m * SST;
        v16u pt;
        #pragma unroll
        for (int j2 = 0; j2 < 8; ++j2) {
            pt[j2]     = f2bf(srow[kbase + j2]);
            pt[j2 + 8] = f2bf(srow[kbase + 16 + j2]);
        }
        v16bf pa = __builtin_bit_cast(v16bf, pt);
        #pragma unroll
        for (int nt = 0; nt < 8; ++nt) {
            v16bf bv = load_fragB_k32(&lv[(nt * 16) * VST + kb * 32], VST);
            o[nt] = wmma_bf16(pa, bv, o[nt]);
        }
    }

    // normalize and write ctx (bf16)
    #pragma unroll
    for (int nt = 0; nt < 8; ++nt) {
        #pragma unroll
        for (int r = 0; r < 8; ++r) {
            int m = r + mh;
            float val = o[nt][r] * linv[wave * 16 + m];
            size_t orow = (size_t)b * SEQ_C + mb * 128 + mw + m;
            ctx[orow * TOTAL_EXP_C + h * 128 + nt * 16 + nl] = f2bf(val);
        }
    }
}

// ---- launch --------------------------------------------------------------

extern "C" void kernel_launch(void* const* d_in, const int* in_sizes, int n_in,
                              void* d_out, int out_size, void* d_ws, size_t ws_size,
                              hipStream_t stream) {
    const float* hs    = (const float*)d_in[0];   // [256,256,448]
    const float* qkvw  = (const float*)d_in[1];   // [448,1536]
    const float* qkvb  = (const float*)d_in[2];   // [1536]
    const float* projw = (const float*)d_in[3];   // [1024,448]
    const float* projb = (const float*)d_in[4];   // [448]
    const float* ab    = (const float*)d_in[5];   // [8,256]
    const int*   idxs  = (const int*)d_in[6];     // [256,256]

    char* ws = (char*)d_ws;
    u16* qkv_ws = (u16*)ws;                                   // 65536*1536 bf16
    size_t off = (size_t)65536 * 1536 * 2;
    u16* ctx_ws = (u16*)(ws + off);                           // 65536*1024 bf16
    off += (size_t)65536 * 1024 * 2;
    float* bias_ws = (float*)(ws + off);                      // 8*65536 f32

    // 1) bias gather
    bias_gather_kernel<<<2048, 256, 0, stream>>>(ab, idxs, bias_ws);

    // 2) QKV = X @ Wqkv + b  (f32 in -> bf16 out)
    gemm_bf16_kernel<false, true><<<dim3(512, 12), 256, 0, stream>>>(
        hs, qkvw, qkvb, qkv_ws, 65536, 1536, 448);

    // 3) fused attention per (batch, head, query-half)
    size_t attn_lds = (size_t)(128 * 32 + 256 * 32 + 128 * VST) * sizeof(u16)
                    + (size_t)(8 * 16 * SST + 8 * 16) * sizeof(float);
    (void)hipFuncSetAttribute((const void*)attn_kernel,
                              hipFuncAttributeMaxDynamicSharedMemorySize,
                              (int)attn_lds);
    attn_kernel<<<256 * 8 * 2, 256, attn_lds, stream>>>(qkv_ws, bias_ws, ctx_ws);

    // 4) out = ctx @ Wproj + b  (bf16 in -> f32 out)
    gemm_bf16_kernel<true, false><<<dim3(512, 4), 256, 0, stream>>>(
        ctx_ws, projw, projb, (float*)d_out, 65536, 448, 1024);
}
